// MultiHeadAttention_16140487098432
// MI455X (gfx1250) — matrix operand order repaired and checked by reading, not compiled, not run on hardware
//
#include <hip/hip_runtime.h>
#include <cstdint>
#include <cstddef>

// ---------------------------------------------------------------------------
// MHA forward for MI455X (gfx1250, wave32, WMMA).
// Sizes hardcoded per reference: B=2, T=2048, C=1024, NH=16, HD=64.
// All GEMM work in bf16 WMMA (f32 accumulate); softmax in fp32.
// GEMMs: 64x64 per-wave tiles (8 b128 A loads + 64 bf16 B loads : 16 WMMAs per 32-K step).
// Attention: flash-style, 64-key chunks per iteration.
// ---------------------------------------------------------------------------

typedef __bf16 bf16;
typedef __bf16 v16bf __attribute__((ext_vector_type(16)));
typedef float  v8f   __attribute__((ext_vector_type(8)));

union Frag {             // one 16x32 (A) or 32x16 (B) bf16 fragment: 8 VGPRs
    v16bf v;
    uint4 q[2];
};

__device__ inline bf16 f2bf(float f) {   // fp32 -> bf16, round-to-nearest-even
    union { float f; uint32_t u; } a; a.f = f;
    uint32_t r = a.u + 0x7FFFu + ((a.u >> 16) & 1u);
    uint16_t h = (uint16_t)(r >> 16);
    union { uint16_t u; bf16 b; } o; o.u = h;
    return o.b;
}

__device__ inline uint4 ld128(const bf16* p) {
    return *reinterpret_cast<const uint4*>(p);
}

__device__ inline v8f wmma_bf16(const Frag& a, const Frag& b, v8f c) {
    // D = A(16x32 bf16) * B(32x16 bf16) + C(16x16 f32)
    return __builtin_amdgcn_wmma_f32_16x16x32_bf16(
        /*neg_a=*/false, a.v, /*neg_b=*/false, b.v,
        /*c_mod=*/(short)0, c, /*reuse_a=*/false, /*reuse_b=*/false);
}

// ---------------------------------------------------------------------------
__global__ void cvt_kernel(const float* __restrict__ in, bf16* __restrict__ out, int n) {
    int i = blockIdx.x * blockDim.x + threadIdx.x;
    if (i < n) out[i] = f2bf(in[i]);
}

// ---------------------------------------------------------------------------
// QKV GEMM: A[4096,1024]bf16 @ W[1024,3072]bf16 + bias -> scatter into
//   Q  [B,NH,T,HD] bf16 (pre-scaled by 1/8 = 1/sqrt(HD))
//   Kt [B,NH,HD,T] bf16 (transposed for contiguous QK^T B-fragments)
//   V  [B,NH,T,HD] bf16
// Per-wave tile 64(M)x64(N); 8 waves in 4(M)x2(N) -> block tile 256x128.
__global__ __launch_bounds__(256)
void qkv_gemm_kernel(const bf16* __restrict__ A, const bf16* __restrict__ W,
                     const float* __restrict__ bias,
                     bf16* __restrict__ qb, bf16* __restrict__ ktb, bf16* __restrict__ vb) {
    const int lane = threadIdx.x & 31;
    const int wave = threadIdx.x >> 5;
    const int half = lane >> 4;       // which 16-lane half
    const int mr   = lane & 15;
    const int m0   = blockIdx.y * 256 + (wave >> 1) * 64;
    const int n0   = blockIdx.x * 128 + (wave & 1) * 64;

    v8f acc[4][4] = {};

    for (int k0 = 0; k0 < 1024; k0 += 32) {
        Frag a[4];
#pragma unroll
        for (int mi = 0; mi < 4; ++mi) {
            // 16-bit A layout: lane<16 row=mr holds K{0..7,16..23}; lane>=16 K{8..15,24..31}
            const bf16* pa = A + (size_t)(m0 + mi * 16 + mr) * 1024 + k0 + half * 8;
            a[mi].q[0] = ld128(pa);
            a[mi].q[1] = ld128(pa + 16);
        }
        Frag b[4];
        const bf16* pb0 = W + (size_t)(k0 + half * 8) * 3072 + n0 + mr; // B layout: lane = N col, K{0..7,16..23} / {8..15,24..31}
#pragma unroll
        for (int tt = 0; tt < 4; ++tt) {
#pragma unroll
            for (int s = 0; s < 16; ++s)
                b[tt].v[s] = pb0[(size_t)((s & 7) + (s >> 3) * 16) * 3072 + tt * 16];
        }
        if (k0 + 128 < 1024) {  // prefetch next weight slab (global_prefetch_b8)
            __builtin_prefetch(W + (size_t)(k0 + 128 + lane) * 3072 + n0, 0, 0);
        }
#pragma unroll
        for (int mi = 0; mi < 4; ++mi)
#pragma unroll
            for (int tt = 0; tt < 4; ++tt)
                acc[mi][tt] = wmma_bf16(a[mi], b[tt], acc[mi][tt]);
    }

    // Epilogue: C-fragment layout is lane -> col (mr), vgpr e -> row (half*8+e)
#pragma unroll
    for (int mi = 0; mi < 4; ++mi) {
#pragma unroll
        for (int tt = 0; tt < 4; ++tt) {
            const int col = n0 + tt * 16 + mr;        // [0,3072)
            const float bs = bias[col];
#pragma unroll
            for (int e = 0; e < 8; ++e) {
                const int row = m0 + mi * 16 + half * 8 + e;  // [0,4096)
                const float val = acc[mi][tt][e] + bs;
                const int bb = row >> 11, t = row & 2047;
                if (col < 1024) {
                    const int h = col >> 6, d = col & 63;
                    qb[(((size_t)(bb * 16 + h)) * 2048 + t) * 64 + d] = f2bf(val * 0.125f);
                } else if (col < 2048) {
                    const int c2 = col - 1024, h = c2 >> 6, d = c2 & 63;
                    ktb[(((size_t)(bb * 16 + h)) * 64 + d) * 2048 + t] = f2bf(val);
                } else {
                    const int c2 = col - 2048, h = c2 >> 6, d = c2 & 63;
                    vb[(((size_t)(bb * 16 + h)) * 2048 + t) * 64 + d] = f2bf(val);
                }
            }
        }
    }
}

// ---------------------------------------------------------------------------
// Flash attention: one wave owns a 16-row query tile; loop over causal 64-key
// chunks. S = Q @ K^T (8 WMMAs), online softmax (fp32), P transposed through
// LDS into A-fragment layout, O += P @ V (8 WMMAs).
__global__ __launch_bounds__(256)
void attn_kernel(const bf16* __restrict__ qb, const bf16* __restrict__ ktb,
                 const bf16* __restrict__ vb, bf16* __restrict__ attn) {
    __shared__ __attribute__((aligned(16))) bf16 Plds[8][16][64]; // per-wave P tile (16KB)

    const int lane = threadIdx.x & 31;
    const int wave = threadIdx.x >> 5;
    const int half = lane >> 4;
    const int mr   = lane & 15;
    const int bh   = blockIdx.y;                      // [0,32) = b*16+h
    const int q0   = (blockIdx.x * 8 + wave) * 16;    // query row base

    const bf16* Q  = qb  + (size_t)bh * 2048 * 64;
    const bf16* Kt = ktb + (size_t)bh * 64 * 2048;
    const bf16* V  = vb  + (size_t)bh * 2048 * 64;

    Frag aQ[2];
#pragma unroll
    for (int c = 0; c < 2; ++c) {   // head-dim chunks 0..31, 32..63
        const bf16* pq = Q + (size_t)(q0 + mr) * 64 + c * 32 + half * 8;
        aQ[c].q[0] = ld128(pq);
        aQ[c].q[1] = ld128(pq + 16);
    }

    v8f accO[4] = {};
    float mrun[8], lrun[8];
#pragma unroll
    for (int e = 0; e < 8; ++e) { mrun[e] = -3.0e38f; lrun[e] = 0.0f; }

    const int jcmax = (q0 + 15) >> 6;
    for (int jc = 0; jc <= jcmax; ++jc) {
        const int j0 = jc * 64;

        // ---- S = Q @ K^T over 4 key tiles of 16 -------------------------
        v8f s[4];
#pragma unroll
        for (int f = 0; f < 4; ++f) {
            Frag bK0, bK1;
            const bf16* pk0 = Kt + (size_t)(half * 8)      * 2048 + j0 + f * 16 + mr;
            const bf16* pk1 = Kt + (size_t)(32 + half * 8) * 2048 + j0 + f * 16 + mr;
#pragma unroll
            for (int s = 0; s < 16; ++s) {
                const size_t ko = (size_t)((s & 7) + (s >> 3) * 16) * 2048;
                bK0.v[s] = pk0[ko]; bK1.v[s] = pk1[ko];
            }
            v8f t = {};
            t = wmma_bf16(aQ[0], bK0, t);
            t = wmma_bf16(aQ[1], bK1, t);
            s[f] = t;
        }

        // ---- causal mask + online softmax (row stats via 16-lane shfl) --
        float p[4][8], sc[8];
#pragma unroll
        for (int e = 0; e < 8; ++e) {
            const int r = q0 + half * 8 + e;
            float cm = -3.0e38f;
#pragma unroll
            for (int f = 0; f < 4; ++f) {
                const float vf = (j0 + f * 16 + mr <= r) ? s[f][e] : -3.0e38f;
                s[f][e] = vf;
                cm = fmaxf(cm, vf);
            }
#pragma unroll
            for (int off = 1; off < 16; off <<= 1)
                cm = fmaxf(cm, __shfl_xor(cm, off, 32));        // row max over 16 cols
            const float mn = fmaxf(mrun[e], cm);
            sc[e] = __expf(mrun[e] - mn);
            mrun[e] = mn;
            float rs = 0.0f;
#pragma unroll
            for (int f = 0; f < 4; ++f) {
                p[f][e] = __expf(s[f][e] - mn);
                rs += p[f][e];
            }
#pragma unroll
            for (int off = 1; off < 16; off <<= 1)
                rs += __shfl_xor(rs, off, 32);                  // row sum
            lrun[e] = lrun[e] * sc[e] + rs;
        }
#pragma unroll
        for (int tt = 0; tt < 4; ++tt)
#pragma unroll
            for (int e = 0; e < 8; ++e) accO[tt][e] *= sc[e];

        // ---- C-layout -> A-layout transpose of P via LDS ----------------
        bf16* pl = &Plds[wave][0][0];                           // [16][64]
#pragma unroll
        for (int f = 0; f < 4; ++f)
#pragma unroll
            for (int e = 0; e < 8; ++e)
                pl[(half * 8 + e) * 64 + f * 16 + mr] = f2bf(p[f][e]);
        asm volatile("s_wait_dscnt 0" ::: "memory");  // wave-local LDS RAW fence

        // ---- O += P @ V over two 32-key sub-chunks ----------------------
#pragma unroll
        for (int kc = 0; kc < 2; ++kc) {
            Frag aP;
            const bf16* pp = pl + mr * 64 + kc * 32 + half * 8;
            aP.q[0] = *reinterpret_cast<const uint4*>(pp);
            aP.q[1] = *reinterpret_cast<const uint4*>(pp + 16);

            const bf16* pv = V + (size_t)(j0 + kc * 32 + half * 8) * 64 + mr;  // lane = value col
#pragma unroll
            for (int tt = 0; tt < 4; ++tt) {
                Frag bV;
#pragma unroll
                for (int s = 0; s < 16; ++s)
                    bV.v[s] = pv[((s & 7) + (s >> 3) * 16) * 64 + tt * 16];
                accO[tt] = wmma_bf16(aP, bV, accO[tt]);
            }
        }
    }

    // Normalize and store to [B,T,C] (head-interleaved) in bf16
    const int b = bh >> 4, h = bh & 15;
#pragma unroll
    for (int tt = 0; tt < 4; ++tt) {
        const int col = h * 64 + tt * 16 + mr;
#pragma unroll
        for (int e = 0; e < 8; ++e) {
            const int t = q0 + half * 8 + e;
            attn[((size_t)(b * 2048 + t)) * 1024 + col] = f2bf(accO[tt][e] / lrun[e]);
        }
    }
}

// ---------------------------------------------------------------------------
// Output projection: attn[4096,1024]bf16 @ Wp[1024,1024]bf16 + bias -> fp32
// Per-wave tile 64x64; block tile 256x128.
__global__ __launch_bounds__(256)
void proj_gemm_kernel(const bf16* __restrict__ A, const bf16* __restrict__ W,
                      const float* __restrict__ bias, float* __restrict__ out) {
    const int lane = threadIdx.x & 31;
    const int wave = threadIdx.x >> 5;
    const int half = lane >> 4;
    const int mr   = lane & 15;
    const int m0   = blockIdx.y * 256 + (wave >> 1) * 64;
    const int n0   = blockIdx.x * 128 + (wave & 1) * 64;

    v8f acc[4][4] = {};

    for (int k0 = 0; k0 < 1024; k0 += 32) {
        Frag a[4];
#pragma unroll
        for (int mi = 0; mi < 4; ++mi) {
            const bf16* pa = A + (size_t)(m0 + mi * 16 + mr) * 1024 + k0 + half * 8;
            a[mi].q[0] = ld128(pa);
            a[mi].q[1] = ld128(pa + 16);
        }
        Frag b[4];
        const bf16* pb0 = W + (size_t)(k0 + half * 8) * 1024 + n0 + mr;
#pragma unroll
        for (int tt = 0; tt < 4; ++tt) {
#pragma unroll
            for (int s = 0; s < 16; ++s)
                b[tt].v[s] = pb0[(size_t)((s & 7) + (s >> 3) * 16) * 1024 + tt * 16];
        }
        if (k0 + 128 < 1024) {  // prefetch next weight slab
            __builtin_prefetch(W + (size_t)(k0 + 128 + lane) * 1024 + n0, 0, 0);
        }
#pragma unroll
        for (int mi = 0; mi < 4; ++mi)
#pragma unroll
            for (int tt = 0; tt < 4; ++tt)
                acc[mi][tt] = wmma_bf16(a[mi], b[tt], acc[mi][tt]);
    }

#pragma unroll
    for (int mi = 0; mi < 4; ++mi) {
#pragma unroll
        for (int tt = 0; tt < 4; ++tt) {
            const int col = n0 + tt * 16 + mr;
            const float bs = bias[col];
#pragma unroll
            for (int e = 0; e < 8; ++e) {
                const int row = m0 + mi * 16 + half * 8 + e;
                out[(size_t)row * 1024 + col] = acc[mi][tt][e] + bs;
            }
        }
    }
}

// ---------------------------------------------------------------------------
extern "C" void kernel_launch(void* const* d_in, const int* in_sizes, int n_in,
                              void* d_out, int out_size, void* d_ws, size_t ws_size,
                              hipStream_t stream) {
    (void)in_sizes; (void)n_in; (void)out_size; (void)ws_size;

    const float* x      = (const float*)d_in[0];   // [2,2048,1024]
    const float* W_qkv  = (const float*)d_in[1];   // [1024,3072]
    const float* b_qkv  = (const float*)d_in[2];   // [3072]
    const float* W_proj = (const float*)d_in[3];   // [1024,1024]
    const float* b_proj = (const float*)d_in[4];   // [1024]
    float* out = (float*)d_out;                    // [2,2048,1024]

    char* w = (char*)d_ws;
    const size_t MB = 1024u * 1024u;
    bf16* xb    = (bf16*)(w + 0 * MB);    // 8 MB  x in bf16 [4096,1024]
    bf16* wqkv  = (bf16*)(w + 8 * MB);    // 6 MB  W_qkv bf16 [1024,3072]
    bf16* wproj = (bf16*)(w + 14 * MB);   // 2 MB  W_proj bf16 [1024,1024]
    bf16* qB    = (bf16*)(w + 16 * MB);   // 8 MB  Q  [32,2048,64] (pre-scaled)
    bf16* ktB   = (bf16*)(w + 24 * MB);   // 8 MB  Kt [32,64,2048]
    bf16* vB    = (bf16*)(w + 32 * MB);   // 8 MB  V  [32,2048,64]
    bf16* attnB = (bf16*)(w + 40 * MB);   // 8 MB  attn out bf16 [4096,1024]

    const int nx = 2 * 2048 * 1024;     // 4194304
    const int nwq = 1024 * 3072;        // 3145728
    const int nwp = 1024 * 1024;        // 1048576
    cvt_kernel<<<(nx  + 255) / 256, 256, 0, stream>>>(x,      xb,    nx);
    cvt_kernel<<<(nwq + 255) / 256, 256, 0, stream>>>(W_qkv,  wqkv,  nwq);
    cvt_kernel<<<(nwp + 255) / 256, 256, 0, stream>>>(W_proj, wproj, nwp);

    qkv_gemm_kernel<<<dim3(24, 16), 256, 0, stream>>>(xb, wqkv, b_qkv, qB, ktB, vB);
    attn_kernel<<<dim3(16, 32), 256, 0, stream>>>(qB, ktB, vB, attnB);
    proj_gemm_kernel<<<dim3(8, 16), 256, 0, stream>>>(attnB, wproj, b_proj, out);
}
